// WYDmLSTM_27891517620877
// MI455X (gfx1250) — compile-verified, matrix operand order checked by reading, not run on hardware
//
#include <hip/hip_runtime.h>
#include <hip/hip_bf16.h>

// ---------- types for WMMA ----------
typedef __attribute__((ext_vector_type(16))) __bf16 v16bf;
typedef __attribute__((ext_vector_type(8)))  float  v8f;
typedef __attribute__((ext_vector_type(2)))  float  v2f;

// pointee type of the async-copy builtins (from hipcc's own diagnostic)
typedef int vi4 __attribute__((vector_size(16)));

#define NTOT 3072            // 6 projections * 512
#define KDIM 512
#define HD   512
#define BB   32
#define SS   2048
#define LDT  40              // LDS row stride (bf16 elems), 80B = 16B aligned, bank-skewed

// CDNA5 async global->LDS copy path (ASYNCcnt), with sync fallback.
#if defined(__gfx1250__) && __has_builtin(__builtin_amdgcn_global_load_async_to_lds_b128) && __has_builtin(__builtin_amdgcn_s_wait_asynccnt)
#define HAVE_ASYNC 1
#else
#define HAVE_ASYNC 0
#endif

// ---------- fp32 <-> bf16 (round-to-nearest-even) ----------
__device__ __forceinline__ __bf16 f2bf(float f) {
  unsigned u = __float_as_uint(f);
  unsigned r = u + 0x7FFFu + ((u >> 16) & 1u);
  unsigned short h = (unsigned short)(r >> 16);
  return __builtin_bit_cast(__bf16, h);
}
__device__ __forceinline__ float bf2f(__bf16 b) {
  unsigned short h = __builtin_bit_cast(unsigned short, b);
  return __uint_as_float(((unsigned)h) << 16);
}

// 16B global->LDS copy: async on CDNA5 (no VGPR staging), else via regs.
__device__ __forceinline__ void copy16_issue(const __bf16* gsrc, __bf16* lds, uint4* stg) {
#if HAVE_ASYNC
  __builtin_amdgcn_global_load_async_to_lds_b128(
      (__attribute__((address_space(1))) vi4*)(__attribute__((address_space(1))) void*)(void*)gsrc,
      (__attribute__((address_space(3))) vi4*)(__attribute__((address_space(3))) void*)(void*)lds,
      0, 0);
  (void)stg;
#else
  *stg = *(const uint4*)gsrc;
#endif
}
__device__ __forceinline__ void copy16_commit(__bf16* lds, const uint4* stg) {
#if !HAVE_ASYNC
  *(uint4*)lds = *stg;
#else
  (void)lds; (void)stg;
#endif
}
__device__ __forceinline__ void async_fence() {
#if HAVE_ASYNC
  __builtin_amdgcn_s_wait_asynccnt(0);
#endif
}

// Load a 16x32 bf16 WMMA A/B fragment from a row-major LDS tile (stride LDT).
// Per ISA: lanes 0-15 -> row = lane&15, K 0..7 & 16..23 ; lanes 16-31 -> K 8..15 & 24..31.
__device__ __forceinline__ v16bf load_frag(const __bf16* tile, int row0) {
  int lane = threadIdx.x & 31;
  int rh   = lane >> 4;
  const __bf16* p = tile + (row0 + (lane & 15)) * LDT;
  union { v16bf v; uint4 q[2]; } f;
  f.q[0] = *(const uint4*)(p + rh * 8);
  f.q[1] = *(const uint4*)(p + 16 + rh * 8);
  return f.v;
}

// =====================================================================
// Phase A: projections GEMM.  C[M, 3072] = X[M,512] @ Wpack^T, fp32 via
// bf16 hi/lo split (3 x v_wmma_f32_16x16x32_bf16), + per-column scale/bias.
// Workgroup tile 64(M) x 128(N), 8 waves in 2x4, each wave 32x32 (4 accs).
// Double-buffered LDS software pipeline; B tiles via async-to-LDS.
// =====================================================================
template<int TC>
__global__ __launch_bounds__(256)
void gemm_proj(const float* __restrict__ x, const __bf16* __restrict__ wh,
               const __bf16* __restrict__ wl, const float* __restrict__ bias,
               float* __restrict__ proj, int s0)
{
  __shared__ __align__(16) __bf16 Ah[2][64 * LDT], Al[2][64 * LDT];
  __shared__ __align__(16) __bf16 Bh[2][128 * LDT], Bl[2][128 * LDT];

  const int tid  = threadIdx.x;
  const int lane = tid & 31, wave = tid >> 5;
  const int wm = wave >> 2, wn = wave & 3;
  const int m0 = blockIdx.y * 64;        // row in chunk (row = b*TC + tl)
  const int n0 = blockIdx.x * 128;

  // A-loader role: 64 rows x 32 k, 8 f32 per thread
  const int ar  = tid >> 2;
  const int akc = (tid & 3) * 8;
  const int agr = m0 + ar;
  const float* asrc = x + (size_t)((agr / TC) * SS + s0 + (agr % TC)) * KDIM + akc;
  const int aoff = ar * LDT + akc;

  // B-loader role: 128 n-rows x 32 k, 16 bf16 (2 x 16B) per thread per matrix
  const int bn   = tid >> 1;
  const int bkc  = (tid & 1) * 16;
  const __bf16* bhsrc = wh + (size_t)(n0 + bn) * KDIM + bkc;
  const __bf16* blsrc = wl + (size_t)(n0 + bn) * KDIM + bkc;
  const int boff = bn * LDT + bkc;

  uint4 stg[4];

  auto storeA = [&](int buf, float4 a0, float4 a1) {
    float v[8] = {a0.x, a0.y, a0.z, a0.w, a1.x, a1.y, a1.z, a1.w};
    union { __bf16 b[8]; uint4 q; } ph, pl;
    #pragma unroll
    for (int i = 0; i < 8; ++i) {
      __bf16 h = f2bf(v[i]);
      ph.b[i] = h;
      pl.b[i] = f2bf(v[i] - bf2f(h));
    }
    *(uint4*)(&Ah[buf][aoff]) = ph.q;
    *(uint4*)(&Al[buf][aoff]) = pl.q;
  };
  auto issueB = [&](int buf, int k0) {
    copy16_issue(bhsrc + k0,     &Bh[buf][boff],     &stg[0]);
    copy16_issue(bhsrc + k0 + 8, &Bh[buf][boff + 8], &stg[1]);
    copy16_issue(blsrc + k0,     &Bl[buf][boff],     &stg[2]);
    copy16_issue(blsrc + k0 + 8, &Bl[buf][boff + 8], &stg[3]);
  };
  auto commitB = [&](int buf) {
    copy16_commit(&Bh[buf][boff],     &stg[0]);
    copy16_commit(&Bh[buf][boff + 8], &stg[1]);
    copy16_commit(&Bl[buf][boff],     &stg[2]);
    copy16_commit(&Bl[buf][boff + 8], &stg[3]);
  };

  v8f acc[2][2] = {};

  // ---- prologue: stage k-tile 0 into buffer 0
  issueB(0, 0);
  {
    float4 a0 = *(const float4*)(asrc);
    float4 a1 = *(const float4*)(asrc + 4);
    storeA(0, a0, a1);
  }
  commitB(0);
  async_fence();
  __syncthreads();

  for (int i = 0; i < KDIM / 32; ++i) {
    const int cur = i & 1, nxt = cur ^ 1;
    const int k1 = (i + 1) * 32;
    const bool more = (i + 1) < (KDIM / 32);

    float4 a0 = {}, a1 = {};
    if (more) {
      issueB(nxt, k1);                       // async engine fills next B tiles
      a0 = *(const float4*)(asrc + k1);      // next A tile in flight
      a1 = *(const float4*)(asrc + k1 + 4);
    }

    v16bf ah[2], al[2], bh[2], bl[2];
    #pragma unroll
    for (int t = 0; t < 2; ++t) {
      ah[t] = load_frag(Ah[cur], wm * 32 + t * 16);
      al[t] = load_frag(Al[cur], wm * 32 + t * 16);
      bh[t] = load_frag(Bh[cur], wn * 32 + t * 16);
      bl[t] = load_frag(Bl[cur], wn * 32 + t * 16);
    }
    #pragma unroll
    for (int mi = 0; mi < 2; ++mi)
      #pragma unroll
      for (int ni = 0; ni < 2; ++ni) {
        acc[mi][ni] = __builtin_amdgcn_wmma_f32_16x16x32_bf16(
            false, ah[mi], false, bh[ni], (short)0, acc[mi][ni], false, false);
        acc[mi][ni] = __builtin_amdgcn_wmma_f32_16x16x32_bf16(
            false, ah[mi], false, bl[ni], (short)0, acc[mi][ni], false, false);
        acc[mi][ni] = __builtin_amdgcn_wmma_f32_16x16x32_bf16(
            false, al[mi], false, bh[ni], (short)0, acc[mi][ni], false, false);
      }

    if (more) {
      storeA(nxt, a0, a1);                   // loadcnt wait lands after WMMAs
      commitB(nxt);
      async_fence();
      __syncthreads();
    }
  }

  // Epilogue: k-projection gets *1/sqrt(H); all get +bias.
  const float inv_sqrt_h = 0.044194173824159216f;
  #pragma unroll
  for (int mi = 0; mi < 2; ++mi)
    #pragma unroll
    for (int ni = 0; ni < 2; ++ni) {
      int ncol = n0 + wn * 32 + ni * 16 + (lane & 15);
      float sc = ((ncol >> 9) == 4) ? inv_sqrt_h : 1.0f;
      float bs = bias[ncol];
      int mbase = m0 + wm * 32 + mi * 16 + ((lane >> 4) << 3);
      #pragma unroll
      for (int v = 0; v < 8; ++v)
        proj[(size_t)(mbase + v) * NTOT + ncol] = acc[mi][ni][v] * sc + bs;
    }
}

// =====================================================================
// Phase B: elementwise recurrence per (b,h) over the chunk's timesteps.
// Emits n_t and g_t = o_t * c_t * q_t (numerator of h_t, pre-denom).
// =====================================================================
template<int TC>
__global__ __launch_bounds__(256)
void scan_chunk(const float* __restrict__ proj, float* __restrict__ mst,
                float* __restrict__ cst, float* __restrict__ nst,
                float* __restrict__ nbuf, float* __restrict__ gbuf)
{
  int tid = blockIdx.x * 256 + threadIdx.x;    // 0..16383
  int b = tid >> 9, h = tid & 511;
  float m = mst[tid], c = cst[tid], n = nst[tid];
  for (int tl = 0; tl < TC; ++tl) {
    const float* p = proj + (size_t)(b * TC + tl) * NTOT + h;
    __builtin_prefetch(p + NTOT, 0, 3);        // next timestep's row
    float it = p[0], ft = p[512], ot = p[1024];
    float qt = p[1536], kt = p[2048], vt = p[2560];
    float o  = 1.0f / (1.0f + __expf(-ot));
    float mt = fmaxf(ft + m, it);
    float ip = __expf(it - mt);
    float fp = __expf(ft + m - mt);
    c = fp * c + ip * (vt * kt);
    n = fp * n + ip * kt;
    m = mt;
    size_t idx = (size_t)(tl * BB + b) * HD + h;
    nbuf[idx] = n;
    gbuf[idx] = o * c * qt;
  }
  mst[tid] = m; cst[tid] = c; nst[tid] = n;
}

// =====================================================================
// Phase C: per-t (parallel over t!) exact fp32 WMMA of D = n_t^T @ q_t,
// denom_j = max_kk |D[j][kk]|, then out[b][t] = fc_b + sum_h g/denom*fcw.
// One workgroup per t; 8 waves; each wave owns 4 j-tiles of 16 rows.
// =====================================================================
template<int TC>
__global__ __launch_bounds__(256)
void denom_out(const float* __restrict__ nbuf, const float* __restrict__ proj,
               const float* __restrict__ gbuf, const float* __restrict__ fcw,
               const float* __restrict__ fcb, float* __restrict__ out, int s0)
{
  __shared__ float denomS[HD];
  const int t    = blockIdx.x;
  const int lane = threadIdx.x & 31, wave = threadIdx.x >> 5;
  const int l15  = lane & 15, rh = lane >> 4;

  const float* nt = nbuf + (size_t)t * BB * HD;   // [b][j]

  #pragma unroll
  for (int jt = 0; jt < 4; ++jt) {
    int j0 = (wave + jt * 8) * 16;
    // A fragments (n^T): A[j][b] = n[b][j]; lane half rh picks K pair {2rh, 2rh+1}.
    v2f afr[8];
    #pragma unroll
    for (int bs = 0; bs < 8; ++bs) {
      int bA = bs * 4 + rh * 2;
      afr[bs].x = nt[(size_t)bA * HD + j0 + l15];
      afr[bs].y = nt[(size_t)(bA + 1) * HD + j0 + l15];
    }
    v8f vmax = {};
    for (int kt = 0; kt < 32; ++kt) {
      int k0 = kt * 16;
      v8f acc = {};
      #pragma unroll
      for (int bs = 0; bs < 8; ++bs) {
        int bB = bs * 4 + rh * 2;
        v2f bfr;
        bfr.x = proj[(size_t)(bB * TC + t) * NTOT + 1536 + k0 + l15];
        bfr.y = proj[(size_t)((bB + 1) * TC + t) * NTOT + 1536 + k0 + l15];
        acc = __builtin_amdgcn_wmma_f32_16x16x4_f32(
            false, afr[bs], false, bfr, (short)0, acc, false, false);
      }
      #pragma unroll
      for (int v = 0; v < 8; ++v) vmax[v] = fmaxf(vmax[v], fabsf(acc[v]));
    }
    // max over the 16 kk lanes of each half (rows differ across halves)
    #pragma unroll
    for (int off = 1; off < 16; off <<= 1)
      #pragma unroll
      for (int v = 0; v < 8; ++v)
        vmax[v] = fmaxf(vmax[v], __shfl_xor(vmax[v], off, 32));
    if (l15 == 0) {
      #pragma unroll
      for (int v = 0; v < 8; ++v) denomS[j0 + rh * 8 + v] = vmax[v];
    }
  }
  __syncthreads();

  float fb = fcb[0];
  #pragma unroll
  for (int bo = 0; bo < 4; ++bo) {
    int b = wave * 4 + bo;
    const float* g = gbuf + (size_t)(t * BB + b) * HD;
    float s = 0.f;
    for (int h = lane; h < HD; h += 32)
      s += g[h] * fcw[h] / denomS[h];
    #pragma unroll
    for (int off = 16; off; off >>= 1) s += __shfl_xor(s, off, 32);
    if (lane == 0) out[(size_t)b * SS + (s0 + t)] = s + fb;
  }
}

// ---------- one-time packing ----------
__global__ void pack_weights(const float* wi, const float* wf, const float* wo,
                             const float* wq, const float* wk, const float* wv,
                             __bf16* wh, __bf16* wl)
{
  int idx = blockIdx.x * 256 + threadIdx.x;     // n*512 + k
  if (idx >= NTOT * KDIM) return;
  int k = idx & 511, n = idx >> 9;
  int h = n & 511, j = n >> 9;
  const float* w = (j == 0) ? wi : (j == 1) ? wf : (j == 2) ? wo
                 : (j == 3) ? wq : (j == 4) ? wk : wv;
  float v = w[(size_t)k * HD + h];              // transpose to [n][k]
  __bf16 hi = f2bf(v);
  wh[idx] = hi;
  wl[idx] = f2bf(v - bf2f(hi));
}

__global__ void init_misc(const float* bi, const float* bf_, const float* bo,
                          const float* bq, const float* bk, const float* bv,
                          float* bias, float* mst, float* cst, float* nst)
{
  int idx = blockIdx.x * 256 + threadIdx.x;
  if (idx < NTOT) {
    int h = idx & 511, j = idx >> 9;
    const float* bb = (j == 0) ? bi : (j == 1) ? bf_ : (j == 2) ? bo
                    : (j == 3) ? bq : (j == 4) ? bk : bv;
    bias[idx] = bb[h];
  }
  if (idx < BB * HD) { mst[idx] = 0.f; cst[idx] = 0.f; nst[idx] = 0.f; }
}

// ---------- host side ----------
template<int TC>
static void run_pipeline(const float* x, const float* const* w, const float* const* bb,
                         const float* fcw, const float* fcb, float* out,
                         char* ws, hipStream_t stream)
{
  size_t off = 0;
  auto alloc = [&](size_t bytes) -> void* {
    void* p = (void*)(ws + off);
    off += (bytes + 255) & ~(size_t)255;
    return p;
  };
  __bf16* wh  = (__bf16*)alloc((size_t)NTOT * KDIM * 2);
  __bf16* wl  = (__bf16*)alloc((size_t)NTOT * KDIM * 2);
  float* bias = (float*) alloc((size_t)NTOT * 4);
  float* mst  = (float*) alloc((size_t)BB * HD * 4);
  float* cst  = (float*) alloc((size_t)BB * HD * 4);
  float* nst  = (float*) alloc((size_t)BB * HD * 4);
  float* proj = (float*) alloc((size_t)BB * TC * NTOT * 4);
  float* nbuf = (float*) alloc((size_t)TC * BB * HD * 4);
  float* gbuf = (float*) alloc((size_t)TC * BB * HD * 4);

  pack_weights<<<(NTOT * KDIM + 255) / 256, 256, 0, stream>>>(
      w[0], w[1], w[2], w[3], w[4], w[5], wh, wl);
  init_misc<<<(BB * HD + 255) / 256, 256, 0, stream>>>(
      bb[0], bb[1], bb[2], bb[3], bb[4], bb[5], bias, mst, cst, nst);

  for (int ch = 0; ch < SS / TC; ++ch) {
    int s0 = ch * TC;
    dim3 ggrid(NTOT / 128, (BB * TC) / 64);
    gemm_proj<TC><<<ggrid, 256, 0, stream>>>(x, wh, wl, bias, proj, s0);
    scan_chunk<TC><<<(BB * HD) / 256, 256, 0, stream>>>(proj, mst, cst, nst, nbuf, gbuf);
    denom_out<TC><<<TC, 256, 0, stream>>>(nbuf, proj, gbuf, fcw, fcb, out, s0);
  }
}

static size_t ws_needed(int TC) {
  size_t s = 0;
  auto add = [&](size_t b) { s += (b + 255) & ~(size_t)255; };
  add((size_t)NTOT * KDIM * 2); add((size_t)NTOT * KDIM * 2); add((size_t)NTOT * 4);
  add((size_t)BB * HD * 4); add((size_t)BB * HD * 4); add((size_t)BB * HD * 4);
  add((size_t)BB * TC * NTOT * 4);
  add((size_t)TC * BB * HD * 4); add((size_t)TC * BB * HD * 4);
  return s;
}

extern "C" void kernel_launch(void* const* d_in, const int* in_sizes, int n_in,
                              void* d_out, int out_size, void* d_ws, size_t ws_size,
                              hipStream_t stream)
{
  (void)in_sizes; (void)n_in; (void)out_size;
  const float* x = (const float*)d_in[0];
  const float* w[6]  = {(const float*)d_in[1], (const float*)d_in[2], (const float*)d_in[3],
                        (const float*)d_in[4], (const float*)d_in[5], (const float*)d_in[6]};
  const float* bb[6] = {(const float*)d_in[7], (const float*)d_in[8], (const float*)d_in[9],
                        (const float*)d_in[10], (const float*)d_in[11], (const float*)d_in[12]};
  const float* fcw = (const float*)d_in[13];
  const float* fcb = (const float*)d_in[14];
  float* out = (float*)d_out;
  char* ws = (char*)d_ws;

  if (ws_size >= ws_needed(256))
    run_pipeline<256>(x, w, bb, fcw, fcb, out, ws, stream);
  else if (ws_size >= ws_needed(64))
    run_pipeline<64>(x, w, bb, fcw, fcb, out, ws, stream);
  else
    run_pipeline<16>(x, w, bb, fcw, fcb, out, ws, stream);
}
